// VectorQuantizer_49134425866694
// MI455X (gfx1250) — matrix operand order repaired and checked by reading, not compiled, not run on hardware
//
#include <hip/hip_runtime.h>

typedef __attribute__((ext_vector_type(16))) __bf16 v16bf;
typedef __attribute__((ext_vector_type(8)))  float  v8f;
typedef __attribute__((ext_vector_type(4)))  float  f4;
typedef __attribute__((ext_vector_type(4)))  unsigned v4u;
typedef __attribute__((ext_vector_type(8)))  int   v8i;
typedef __attribute__((ext_vector_type(4)))  int   v4i;

#define SEGS    8
#define EDIM    64
#define KCODES  1024
#define NVEC    16384      // B*H*W = 16*32*32
#define TILE_N  128
#define ZROW    132        // padded LDS row stride (floats) = 128 data + 4 pad

__device__ __forceinline__ unsigned short bf_bits(float f) {
  unsigned u = __float_as_uint(f);
  return (unsigned short)((u + 0x7FFFu + ((u >> 16) & 1u)) >> 16);   // RNE
}
__device__ __forceinline__ __bf16 f2bf(float f) {
  unsigned short b = bf_bits(f);
  return __builtin_bit_cast(__bf16, b);
}
__device__ __forceinline__ float bfb2f(unsigned short b) {
  return __uint_as_float(((unsigned)b) << 16);
}

// ---------------------------------------------------------------------------
// Prep: split codebooks into bf16 hi/lo planes + exact f32 row norms.
// 8192 rows, 4 threads/row, 16 elems/thread. grid = 128 x 256.
// ---------------------------------------------------------------------------
__global__ __launch_bounds__(256) void vq_prep(
    const float* __restrict__ emb, __bf16* __restrict__ ehi,
    __bf16* __restrict__ elo, float* __restrict__ norms) {
  int gt   = blockIdx.x * 256 + threadIdx.x;
  int row  = gt >> 2;
  int part = gt & 3;
  const float* rp = emb + (size_t)row * EDIM + part * 16;
  __bf16* hp = ehi + (size_t)row * EDIM + part * 16;
  __bf16* lp = elo + (size_t)row * EDIM + part * 16;
  float acc = 0.f;
#pragma unroll
  for (int q = 0; q < 4; ++q) {
    f4 v = *(const f4*)(rp + q * 4);
#pragma unroll
    for (int j = 0; j < 4; ++j) {
      float x = v[j];
      acc = fmaf(x, x, acc);
      unsigned short hb = bf_bits(x);
      hp[q * 4 + j] = __builtin_bit_cast(__bf16, hb);
      lp[q * 4 + j] = f2bf(x - bfb2f(hb));
    }
  }
  acc += __shfl_xor(acc, 1, 4);
  acc += __shfl_xor(acc, 2, 4);
  if (part == 0) norms[row] = acc;
}

// ---------------------------------------------------------------------------
// Main: per-WG = one segment s x 128 vectors. 8 waves; wave w owns rows
// w*16..w*16+15. TDM tile load -> LDS, bf16x2-split WMMA distance GEMM +
// fused argmin + gather.
// ---------------------------------------------------------------------------
__global__ __launch_bounds__(256) void vq_main(
    const float* __restrict__ z, const float* __restrict__ embf,
    const __bf16* __restrict__ ehi, const __bf16* __restrict__ elo,
    const float* __restrict__ norms, float* __restrict__ out,
    float* __restrict__ partials) {
  __shared__ float zt[EDIM * ZROW];   // z tile, f32, [e][n], 132-dword rows
  __shared__ int   idx_s[TILE_N];
  __shared__ float red[256];

  const int bx   = blockIdx.x;
  const int s    = bx >> 7;           // 128 tiles per segment
  const int tile = bx & 127;
  const int n_base = tile * TILE_N;   // tiles never cross batch boundary
  const int b  = n_base >> 10;        // H*W = 1024
  const int hw = n_base & 1023;
  const int tid = threadIdx.x;
  const int lane = tid & 31;
  const int wv   = tid >> 5;

  const float* zp = z + ((size_t)b * 512 + (size_t)s * EDIM) * 1024 + hw;

  // ---- stage 1: z tile -> LDS ----
#if __has_builtin(__builtin_amdgcn_tensor_load_to_lds)
  // Tensor Data Mover: one DMA moves the 64x128 f32 tile (row stride 1024 f32)
  // into LDS with hardware padding (128 dwords + 4 pad = ZROW). Wave 0 issues
  // (TDM ignores EXEC; issuing from all 8 waves would duplicate the DMA).
  if (wv == 0) {
    unsigned ldsoff = (unsigned)(size_t)(__attribute__((address_space(3))) float*)zt;
    unsigned long long ga = (unsigned long long)(const void*)zp;
    v4u g0;
    g0[0] = 1u;                                      // count=1 (valid user D#)
    g0[1] = ldsoff;                                  // lds_addr (bytes)
    g0[2] = (unsigned)(ga & 0xFFFFFFFFu);            // global_addr[31:0]
    g0[3] = (unsigned)((ga >> 32) & 0x01FFFFFFu)     // global_addr[56:32]
          | (2u << 30);                              // type = 2 ("image")
    v8i g1;
    g1[0] = (int)((2u << 16)    // data_size = 2 -> 4 bytes
                | (1u << 20)    // pad_enable
                | (6u << 22)    // pad_interval: 128 dwords between pads
                | (3u << 25));  // pad_amount:   4 dwords of pad
    g1[1] = (int)(128u << 16);  // tensor_dim0 = 128 (x-extent from tile start)
    g1[2] = (int)(64u  << 16);  // tensor_dim0[31:16]=0 | tensor_dim1 = 64
    g1[3] = (int)(128u << 16);  // tensor_dim1[31:16]=0 | tile_dim0 = 128
    g1[4] = 64;                 // tile_dim1 = 64 | tile_dim2 = 0
    g1[5] = 1024;               // tensor_dim0_stride = 1024 elements
    g1[6] = 0;                  // stride0[47:32]=0 | tensor_dim1_stride lo = 0
    g1[7] = 0;                  // tensor_dim1_stride hi = 0
    v4i gz = {0, 0, 0, 0};      // 2D tile: groups 2/3 unused
#if defined(__clang_major__) && __clang_major__ >= 23
    v8i gz8 = {0, 0, 0, 0, 0, 0, 0, 0};
    __builtin_amdgcn_tensor_load_to_lds(g0, g1, gz, gz, gz8, 0);
#else
    __builtin_amdgcn_tensor_load_to_lds(g0, g1, gz, gz, 0);
#endif
    __builtin_amdgcn_s_wait_tensorcnt(0);
  }
  __syncthreads();
#else
  // Fallback: coalesced VMEM load + LDS store.
#pragma unroll
  for (int it = 0; it < 8; ++it) {
    int flat = it * 256 + tid;        // 2048 float4 groups
    int e  = flat >> 5;
    int n4 = (flat & 31) << 2;
    f4 v = *(const f4*)(zp + (size_t)e * 1024 + n4);
    *(f4*)&zt[e * ZROW + n4] = v;
  }
  __syncthreads();
#endif

  const int hf = lane >> 4;           // lane half (0/1)
  const int m  = lane & 15;           // A-matrix row / D column
  const int nl = wv * 16 + m;         // this lane's n_local

  // ---- A fragments (hi/lo), built once, 16-bit A 16x32 layout ----
  v16bf a0h, a0l, a1h, a1l;
#pragma unroll
  for (int i = 0; i < 8; ++i) {
    int e = hf * 16 + i;
    float x0 = zt[(e)      * ZROW + nl];
    float x1 = zt[(e + 8)  * ZROW + nl];
    float x2 = zt[(e + 32) * ZROW + nl];
    float x3 = zt[(e + 40) * ZROW + nl];
    unsigned short h0 = bf_bits(x0), h1 = bf_bits(x1);
    unsigned short h2 = bf_bits(x2), h3 = bf_bits(x3);
    a0h[i]     = __builtin_bit_cast(__bf16, h0);
    a0h[i + 8] = __builtin_bit_cast(__bf16, h1);
    a1h[i]     = __builtin_bit_cast(__bf16, h2);
    a1h[i + 8] = __builtin_bit_cast(__bf16, h3);
    a0l[i]     = f2bf(x0 - bfb2f(h0));
    a0l[i + 8] = f2bf(x1 - bfb2f(h1));
    a1l[i]     = f2bf(x2 - bfb2f(h2));
    a1l[i + 8] = f2bf(x3 - bfb2f(h3));
  }

  // ---- stage 2: K loop, 16 codes per chunk, fused argmin ----
  float minv[8];
  int   mini[8];
#pragma unroll
  for (int r = 0; r < 8; ++r) { minv[r] = 3.4e38f; mini[r] = 0; }

  const int kc = m;                   // this lane's code column
  const __bf16* rh0 = ehi + ((size_t)s * KCODES + kc) * EDIM;
  const __bf16* rl0 = elo + ((size_t)s * KCODES + kc) * EDIM;

  for (int kb = 0; kb < KCODES; kb += 16) {
    float nk = norms[s * KCODES + kb + kc];
    const __bf16* rh = rh0 + (size_t)kb * EDIM;
    const __bf16* rl = rl0 + (size_t)kb * EDIM;
    v16bf b0h = *(const v16bf*)(rh + hf * 16);
    v16bf b1h = *(const v16bf*)(rh + 32 + hf * 16);
    v16bf b0l = *(const v16bf*)(rl + hf * 16);
    v16bf b1l = *(const v16bf*)(rl + 32 + hf * 16);
    if (kb + 16 < KCODES) {           // hint next chunk into cache
      __builtin_prefetch(rh + 16 * EDIM, 0, 3);
      __builtin_prefetch(rl + 16 * EDIM, 0, 3);
    }
    v8f c = {};
    // low-order terms first, hi*hi last (bf16x2 split => ~fp32 dot)
    c = __builtin_amdgcn_wmma_f32_16x16x32_bf16(false, a0l, false, b0h, (short)0, c, false, false);
    c = __builtin_amdgcn_wmma_f32_16x16x32_bf16(false, a0h, false, b0l, (short)0, c, false, false);
    c = __builtin_amdgcn_wmma_f32_16x16x32_bf16(false, a1l, false, b1h, (short)0, c, false, false);
    c = __builtin_amdgcn_wmma_f32_16x16x32_bf16(false, a1h, false, b1l, (short)0, c, false, false);
    c = __builtin_amdgcn_wmma_f32_16x16x32_bf16(false, a0h, false, b0h, (short)0, c, false, false);
    c = __builtin_amdgcn_wmma_f32_16x16x32_bf16(false, a1h, false, b1h, (short)0, c, false, false);
    int k = kb + kc;
#pragma unroll
    for (int r = 0; r < 8; ++r) {
      float score = fmaf(-2.f, c[r], nk);   // ||e||^2 - 2 z.e  (||z||^2 const)
      bool lt = score < minv[r];            // strict: keeps earliest k
      minv[r] = lt ? score : minv[r];
      mini[r] = lt ? k : mini[r];
    }
  }

  // ---- argmin reduction across the 16 code-columns of each half-wave ----
#pragma unroll
  for (int off = 1; off < 16; off <<= 1) {
#pragma unroll
    for (int r = 0; r < 8; ++r) {
      float ov = __shfl_xor(minv[r], off, 16);
      int   oi = __shfl_xor(mini[r], off, 16);
      if (ov < minv[r] || (ov == minv[r] && oi < mini[r])) {
        minv[r] = ov; mini[r] = oi;
      }
    }
  }
  if (m == 0) {                 // lanes 0 (rows 0-7) and 16 (rows 8-15)
#pragma unroll
    for (int r = 0; r < 8; ++r) idx_s[wv * 16 + hf * 8 + r] = mini[r];
  }
  __syncthreads();

  // ---- stage 3: gather codes, write zq (coalesced along n), fused loss ----
  float lsum = 0.f;
  float* outq = out + 1;        // d_out[0] = loss, rest = z_q NCHW
  const size_t obase = ((size_t)b * 512 + (size_t)s * EDIM) * 1024 + hw;
#pragma unroll
  for (int it = 0; it < 8; ++it) {
    int flat = it * 256 + tid;
    int e  = flat >> 5;
    int n4 = (flat & 31) << 2;
    size_t oa = obase + (size_t)e * 1024 + n4;
#pragma unroll
    for (int j = 0; j < 4; ++j) {
      int idxv = idx_s[n4 + j];
      float q  = embf[((size_t)s * KCODES + idxv) * EDIM + e];  // exact f32 code
      float zf = zt[e * ZROW + n4 + j];
      float d  = q - zf;
      lsum = fmaf(d, d, lsum);
      outq[oa + j] = q;         // straight-through forward value == zq
    }
  }
  red[tid] = lsum;
  __syncthreads();
  for (int st = 128; st > 0; st >>= 1) {       // fixed-order => deterministic
    if (tid < st) red[tid] += red[tid + st];
    __syncthreads();
  }
  if (tid == 0) partials[bx] = red[0];
}

// ---------------------------------------------------------------------------
// Finalize: deterministic sum of per-WG partials, scale, write loss.
// loss = (1 + BETA) * sum_s mean_{n,e} diff^2 = 1.25 * total / (16384*64)
// ---------------------------------------------------------------------------
__global__ __launch_bounds__(256) void vq_final(
    const float* __restrict__ partials, float* __restrict__ out) {
  __shared__ float red[256];
  int tid = threadIdx.x;
  float s = 0.f;
  for (int i = tid; i < 1024; i += 256) s += partials[i];
  red[tid] = s;
  __syncthreads();
  for (int st = 128; st > 0; st >>= 1) {
    if (tid < st) red[tid] += red[tid + st];
    __syncthreads();
  }
  if (tid == 0) out[0] = red[0] * (1.25f / 1048576.f);
}

extern "C" void kernel_launch(void* const* d_in, const int* in_sizes, int n_in,
                              void* d_out, int out_size, void* d_ws, size_t ws_size,
                              hipStream_t stream) {
  (void)in_sizes; (void)n_in; (void)out_size; (void)ws_size;
  const float* z   = (const float*)d_in[0];   // [16,512,32,32] f32
  const float* emb = (const float*)d_in[1];   // [8,1024,64]    f32
  float* out = (float*)d_out;                 // [1 + 16*512*32*32] f32
  char*  ws  = (char*)d_ws;

  __bf16* ehi   = (__bf16*)(ws);                          // 1 MB
  __bf16* elo   = (__bf16*)(ws + (size_t)(1u << 20));     // 1 MB
  float*  norms = (float*)(ws + (size_t)(2u << 20));      // 32 KB
  float*  parts = (float*)(ws + (size_t)(2u << 20) + 32768); // 4 KB

  hipLaunchKernelGGL(vq_prep,  dim3(128),  dim3(256), 0, stream, emb, ehi, elo, norms);
  hipLaunchKernelGGL(vq_main,  dim3(1024), dim3(256), 0, stream, z, emb, ehi, elo,
                     norms, out, parts);
  hipLaunchKernelGGL(vq_final, dim3(1),    dim3(256), 0, stream, parts, out);
}
